// EncoderBlock_87582973100769
// MI455X (gfx1250) — matrix operand order repaired and checked by reading, not compiled, not run on hardware
//
#include <hip/hip_runtime.h>

#define B_  2
#define S_  2048
#define D_  1024
#define H_  16
#define DK_ 64
#define FF_ 4096
#define M_  (B_ * S_)      // 4096 token rows
#define EPS_ 1e-5f
#define NEGBIG (-1e30f)

typedef __bf16 bf16;
typedef __attribute__((ext_vector_type(16))) __bf16 v16bf;
typedef __attribute__((ext_vector_type(8)))  __bf16 v8bf;
typedef __attribute__((ext_vector_type(8)))  float  v8f;
typedef __attribute__((ext_vector_type(4)))  unsigned int ui32x4;
typedef __attribute__((ext_vector_type(8)))  int i32x8;
typedef __attribute__((ext_vector_type(4)))  int i32x4;

static __device__ __forceinline__ v8f wmma_bf16(v16bf a, v16bf b, v8f c) {
  // D = A(16x32 bf16) * B(32x16 bf16) + C(16x16 f32)
  return __builtin_amdgcn_wmma_f32_16x16x32_bf16(false, a, false, b, (short)0, c,
                                                 false, false);
}

static __device__ __forceinline__ v8f v8f_zero() {
  v8f z = {0.f, 0.f, 0.f, 0.f, 0.f, 0.f, 0.f, 0.f};
  return z;
}

// A-fragment 16x32 bf16 (row-major source, leading dim ld, base already at tile):
// lane L -> row (L&15); K range [half*16 .. +15] contiguous, half=L>>4.
static __device__ __forceinline__ v16bf load_afrag(const bf16* base, int ld, int lane) {
  const bf16* p = base + (long)(lane & 15) * ld + ((lane >> 4) << 4);
  v8bf lo = *(const v8bf*)p;
  v8bf hi = *(const v8bf*)(p + 8);
  v16bf r;
#pragma unroll
  for (int i = 0; i < 8; ++i) { r[i] = lo[i]; r[i + 8] = hi[i]; }
  return r;
}

// B-fragment 32x16 bf16 where B = Wt (W stored row-major [N x K]):
// lane L -> column n=(L&15) of B == row n of W; K range [half*16 .. +15] contiguous.
static __device__ __forceinline__ v16bf load_bfrag_rowmajor(const bf16* base, int ld,
                                                            int lane) {
  const bf16* p = base + (long)(lane & 15) * ld + ((lane >> 4) << 4);
  v8bf lo = *(const v8bf*)p;
  v8bf hi = *(const v8bf*)(p + 8);
  v16bf r;
#pragma unroll
  for (int i = 0; i < 8; ++i) { r[i] = lo[i]; r[i + 8] = hi[i]; }
  return r;
}

// ---------------- Tensor Data Mover (TDM) helpers -------------------------
// D# descriptor per CDNA5 ISA. 2-D tile load:
//   group0: count=1, lds_addr[63:32], global_addr[120:64], type=2 [127:126]
//   group1: data_size (2B), tensor_dim0/1, tile_dim0/1, tensor_dim0_stride

static __device__ __forceinline__ unsigned int lds_offset(const void* p) {
  return (unsigned int)(unsigned long long)(const __attribute__((address_space(3))) void*)p;
}

static __device__ __forceinline__ i32x8 pack_q4(unsigned long long q0,
                                                unsigned long long q1,
                                                unsigned long long q2,
                                                unsigned long long q3) {
  i32x8 v;
  v[0] = (int)(unsigned int)q0; v[1] = (int)(unsigned int)(q0 >> 32);
  v[2] = (int)(unsigned int)q1; v[3] = (int)(unsigned int)(q1 >> 32);
  v[4] = (int)(unsigned int)q2; v[5] = (int)(unsigned int)(q2 >> 32);
  v[6] = (int)(unsigned int)q3; v[7] = (int)(unsigned int)(q3 >> 32);
  return v;
}

// Load a [tile_rows x tile_k] bf16 tile (row-major, row stride = stride_elems)
// from global into LDS at lds_byte (rows packed contiguously, ld = tile_k).
static __device__ __forceinline__ void tdm_load_2d(const void* gaddr,
                                                   unsigned int lds_byte,
                                                   unsigned int tile_k,
                                                   unsigned int tile_rows,
                                                   unsigned long long stride_elems) {
  const unsigned long long ga = (unsigned long long)gaddr;
  ui32x4 g0;
  g0[0] = 1u;                                              // count=1 (valid user D#)
  g0[1] = lds_byte;                                        // lds_addr
  g0[2] = (unsigned int)ga;                                // global_addr[31:0]
  g0[3] = (unsigned int)((ga >> 32) & 0x1FFFFFFu) | (2u << 30);  // [56:32] + type=2
  // group1 bitfields (bit positions are within the 256-bit group):
  //  [17:16]=data_size(1->2B)  [79:48]=tensor_dim0  [111:80]=tensor_dim1
  //  [127:112]=tile_dim0  [143:128]=tile_dim1  [207:160]=tensor_dim0_stride
  const unsigned long long q0 =
      (1ull << 16) | ((unsigned long long)(tile_k & 0xFFFFu) << 48);
  const unsigned long long q1 =
      (unsigned long long)(tile_k >> 16) |
      ((unsigned long long)tile_rows << 16) |
      ((unsigned long long)(tile_k & 0xFFFFu) << 48);
  const unsigned long long q2 =
      (unsigned long long)(tile_rows & 0xFFFFu) |
      ((stride_elems & 0xFFFFFFFFull) << 32);
  const unsigned long long q3 = (stride_elems >> 32) & 0xFFFFull;
  const i32x8 g1 = pack_q4(q0, q1, q2, q3);
  const i32x4 z4 = {0, 0, 0, 0};
  const i32x8 z8 = {0, 0, 0, 0, 0, 0, 0, 0};
  // amdgpu-toolchain (clang-23) 6-arg form; emits tensor_load_to_lds (TENSORcnt).
  __builtin_amdgcn_tensor_load_to_lds(g0, g1, z4, z4, z8, 0);
}

template <bool RELU, bool RES, bool OUTBF>
static __device__ __forceinline__ void epilogue(v8f acc, int r, int c, int N,
                                                const float* __restrict__ bias,
                                                const float* __restrict__ res,
                                                float* __restrict__ outf,
                                                bf16* __restrict__ outb) {
  const float bv = bias[c];
#pragma unroll
  for (int i = 0; i < 8; ++i) {
    float v = acc[i] + bv;
    if (RELU) v = fmaxf(v, 0.0f);
    const long idx = (long)(r + i) * N + c;
    if (RES) v += res[idx];
    if (OUTBF) outb[idx] = (bf16)v;
    else       outf[idx] = v;
  }
}

// C[M,N] = A[M,K](bf16) * W[N,K]^T(bf16) + bias (+res) (relu?) -> f32 or bf16
// Block: 256 threads = 8 waves. Block tile 128x128, K-slab 64, double-buffered
// LDS (64 KB) filled by the Tensor Data Mover from wave 0, synchronized with
// s_wait_tensorcnt + workgroup barrier. Wave tile 64x32 = 4x2 WMMA accums.
template <bool RELU, bool RES, bool OUTBF>
__global__ __launch_bounds__(256) void gemm_bf16_kernel(
    const bf16* __restrict__ A, const bf16* __restrict__ W,
    const float* __restrict__ bias, const float* __restrict__ res,
    float* __restrict__ outf, bf16* __restrict__ outb, int M, int N, int K) {
  __shared__ __align__(16) bf16 sA[2][128 * 64];
  __shared__ __align__(16) bf16 sB[2][128 * 64];

  const int lane = threadIdx.x & 31;
  const int wave = threadIdx.x >> 5;
  const int wm = wave >> 2;      // 0..1 -> 64-row strip
  const int wn = wave & 3;       // 0..3 -> 32-col strip

  const bf16* Ablk = A + (long)(blockIdx.y * 128) * K;
  const bf16* Wblk = W + (long)(blockIdx.x * 128) * K;

  v8f acc[4][2];
#pragma unroll
  for (int t = 0; t < 4; ++t)
#pragma unroll
    for (int u = 0; u < 2; ++u) acc[t][u] = v8f_zero();

  const int nslab = K >> 6;   // K / 64
  if (wave == 0) {
    tdm_load_2d(Ablk, lds_offset(&sA[0][0]), 64, 128, (unsigned long long)K);
    tdm_load_2d(Wblk, lds_offset(&sB[0][0]), 64, 128, (unsigned long long)K);
  }

  for (int i = 0; i < nslab; ++i) {
    if (wave == 0) {
      if (i + 1 < nslab) {
        const int k1 = (i + 1) << 6;
        tdm_load_2d(Ablk + k1, lds_offset(&sA[(i + 1) & 1][0]), 64, 128,
                    (unsigned long long)K);
        tdm_load_2d(Wblk + k1, lds_offset(&sB[(i + 1) & 1][0]), 64, 128,
                    (unsigned long long)K);
        __builtin_amdgcn_s_wait_tensorcnt(2);  // slab i complete (in-order)
      } else {
        __builtin_amdgcn_s_wait_tensorcnt(0);
      }
    }
    __syncthreads();

    const bf16* At = &sA[i & 1][0];
    const bf16* Bt = &sB[i & 1][0];
#pragma unroll
    for (int kk = 0; kk < 64; kk += 32) {
      v16bf a0 = load_afrag(At + (wm * 64 +  0) * 64 + kk, 64, lane);
      v16bf a1 = load_afrag(At + (wm * 64 + 16) * 64 + kk, 64, lane);
      v16bf a2 = load_afrag(At + (wm * 64 + 32) * 64 + kk, 64, lane);
      v16bf a3 = load_afrag(At + (wm * 64 + 48) * 64 + kk, 64, lane);
      v16bf b0 = load_bfrag_rowmajor(Bt + (wn * 32 +  0) * 64 + kk, 64, lane);
      v16bf b1 = load_bfrag_rowmajor(Bt + (wn * 32 + 16) * 64 + kk, 64, lane);
      acc[0][0] = wmma_bf16(a0, b0, acc[0][0]);
      acc[0][1] = wmma_bf16(a0, b1, acc[0][1]);
      acc[1][0] = wmma_bf16(a1, b0, acc[1][0]);
      acc[1][1] = wmma_bf16(a1, b1, acc[1][1]);
      acc[2][0] = wmma_bf16(a2, b0, acc[2][0]);
      acc[2][1] = wmma_bf16(a2, b1, acc[2][1]);
      acc[3][0] = wmma_bf16(a3, b0, acc[3][0]);
      acc[3][1] = wmma_bf16(a3, b1, acc[3][1]);
    }
    __syncthreads();
  }

  const int cn = lane & 15;
  const int rm = (lane >> 4) * 8;
  const int row0 = blockIdx.y * 128 + wm * 64 + rm;
  const int col0 = blockIdx.x * 128 + wn * 32 + cn;
#pragma unroll
  for (int t = 0; t < 4; ++t) {
    epilogue<RELU, RES, OUTBF>(acc[t][0], row0 + t * 16, col0,      N, bias, res, outf, outb);
    epilogue<RELU, RES, OUTBF>(acc[t][1], row0 + t * 16, col0 + 16, N, bias, res, outf, outb);
  }
}

// Flash-style attention: one wave handles 16 query rows x DK=64, streaming keys
// in slabs of 32 with online softmax. Q/K/V layout: [B*S, D] with head h at
// columns h*64..h*64+63. Output O written bf16 in the same layout.
__global__ __launch_bounds__(256) void attn_kernel(
    const bf16* __restrict__ Q, const bf16* __restrict__ Kmat,
    const bf16* __restrict__ V, const int* __restrict__ mask,
    bf16* __restrict__ O) {
  __shared__ __align__(16) bf16 sP[8][16 * 32];  // per-wave P-tile scratch (8 KB)
  const int lane = threadIdx.x & 31;
  const int wave = threadIdx.x >> 5;
  const int b = blockIdx.x >> 4;   // H_ == 16
  const int h = blockIdx.x & 15;
  const int q0 = (blockIdx.y * 8 + wave) * 16;
  const int half = lane >> 4;
  const int n = lane & 15;

  const bf16* Qb = Q    + (long)(b * S_ + q0) * D_ + h * DK_;
  const bf16* Kb = Kmat + (long)(b * S_) * D_ + h * DK_;
  const bf16* Vb = V    + (long)(b * S_) * D_ + h * DK_;
  const int*  mk = mask + b * S_;

  const v16bf qa0 = load_afrag(Qb, D_, lane);
  const v16bf qa1 = load_afrag(Qb + 32, D_, lane);

  float mrun[8], lrun[8];
#pragma unroll
  for (int i = 0; i < 8; ++i) { mrun[i] = NEGBIG; lrun[i] = 0.0f; }
  v8f o[4];
#pragma unroll
  for (int j = 0; j < 4; ++j) o[j] = v8f_zero();

  for (int kb = 0; kb < S_; kb += 32) {
    v16bf kf00 = load_bfrag_rowmajor(Kb + (long)kb * D_,             D_, lane);
    v16bf kf01 = load_bfrag_rowmajor(Kb + (long)kb * D_ + 32,        D_, lane);
    v16bf kf10 = load_bfrag_rowmajor(Kb + (long)(kb + 16) * D_,      D_, lane);
    v16bf kf11 = load_bfrag_rowmajor(Kb + (long)(kb + 16) * D_ + 32, D_, lane);
    v8f s0 = v8f_zero(), s1 = v8f_zero();
    s0 = wmma_bf16(qa0, kf00, s0);
    s0 = wmma_bf16(qa1, kf01, s0);
    s1 = wmma_bf16(qa0, kf10, s1);
    s1 = wmma_bf16(qa1, kf11, s1);

    const float msk0 = (mk[kb + n] == 0)      ? NEGBIG : 0.0f;
    const float msk1 = (mk[kb + 16 + n] == 0) ? NEGBIG : 0.0f;

    float p0[8], p1[8], alpha[8];
#pragma unroll
    for (int i = 0; i < 8; ++i) {
      float a = s0[i] * 0.125f + msk0;   // 1/sqrt(64)
      float c = s1[i] * 0.125f + msk1;
      float mt = fmaxf(a, c);
      mt = fmaxf(mt, __shfl_xor(mt, 1, 32));
      mt = fmaxf(mt, __shfl_xor(mt, 2, 32));
      mt = fmaxf(mt, __shfl_xor(mt, 4, 32));
      mt = fmaxf(mt, __shfl_xor(mt, 8, 32));
      const float mn = fmaxf(mrun[i], mt);
      const float al = __expf(mrun[i] - mn);
      const float e0 = __expf(a - mn);
      const float e1 = __expf(c - mn);
      float ls = e0 + e1;
      ls += __shfl_xor(ls, 1, 32);
      ls += __shfl_xor(ls, 2, 32);
      ls += __shfl_xor(ls, 4, 32);
      ls += __shfl_xor(ls, 8, 32);
      lrun[i] = lrun[i] * al + ls;
      mrun[i] = mn;
      alpha[i] = al; p0[i] = e0; p1[i] = e1;
    }
#pragma unroll
    for (int j = 0; j < 4; ++j)
#pragma unroll
      for (int i = 0; i < 8; ++i) o[j][i] *= alpha[i];

    // C-layout -> A-layout for P via per-wave LDS (same-wave DS is in-order)
    bf16* sp = &sP[wave][0];
#pragma unroll
    for (int i = 0; i < 8; ++i) {
      sp[(i + half * 8) * 32 + n]      = (bf16)p0[i];
      sp[(i + half * 8) * 32 + 16 + n] = (bf16)p1[i];
    }
    __builtin_amdgcn_wave_barrier();
    v16bf pa = load_afrag(sp, 32, lane);
    __builtin_amdgcn_wave_barrier();

#pragma unroll
    for (int j = 0; j < 4; ++j) {
      const bf16* vp = Vb + (long)(kb + half * 16) * D_ + j * 16 + n;
      v16bf vf;
#pragma unroll
      for (int e = 0; e < 16; ++e) vf[e] = vp[(long)e * D_];
      o[j] = wmma_bf16(pa, vf, o[j]);
    }
  }

  bf16* Ob = O + (long)(b * S_ + q0 + half * 8) * D_ + h * DK_;
#pragma unroll
  for (int j = 0; j < 4; ++j)
#pragma unroll
    for (int i = 0; i < 8; ++i)
      Ob[(long)i * D_ + j * 16 + n] = (bf16)(o[j][i] / lrun[i]);
}

// LayerNorm matching reference: mean, std(ddof=1), (x-mean)/(std+eps)*a+b -> bf16
__global__ __launch_bounds__(256) void ln_kernel(const float* __restrict__ x,
                                                 const float* __restrict__ alpha,
                                                 const float* __restrict__ beta,
                                                 bf16* __restrict__ out) {
  __shared__ float rs[8], rq[8];
  const long row = blockIdx.x;
  const float* xr = x + row * D_;
  const int lane = threadIdx.x & 31;
  const int wave = threadIdx.x >> 5;
  float s = 0.f, q = 0.f;
  for (int i = threadIdx.x; i < D_; i += 256) {
    const float v = xr[i];
    s += v;
    q += v * v;
  }
#pragma unroll
  for (int d = 1; d < 32; d <<= 1) {
    s += __shfl_xor(s, d, 32);
    q += __shfl_xor(q, d, 32);
  }
  if (lane == 0) { rs[wave] = s; rq[wave] = q; }
  __syncthreads();
  float ts = 0.f, tq = 0.f;
#pragma unroll
  for (int w = 0; w < 8; ++w) { ts += rs[w]; tq += rq[w]; }
  const float mean = ts * (1.0f / D_);
  const float var = (tq - (float)D_ * mean * mean) * (1.0f / (D_ - 1));
  const float inv = 1.0f / (sqrtf(fmaxf(var, 0.f)) + EPS_);
  bf16* orow = out + row * D_;
  for (int i = threadIdx.x; i < D_; i += 256)
    orow[i] = (bf16)((xr[i] - mean) * inv * alpha[i] + beta[i]);
}

__global__ __launch_bounds__(256) void cvt_kernel(const float* __restrict__ in,
                                                  bf16* __restrict__ out, int nelem) {
  for (int i = blockIdx.x * 256 + threadIdx.x; i < nelem; i += gridDim.x * 256)
    out[i] = (bf16)in[i];
}

extern "C" void kernel_launch(void* const* d_in, const int* in_sizes, int n_in,
                              void* d_out, int out_size, void* d_ws, size_t ws_size,
                              hipStream_t stream) {
  const float* x     = (const float*)d_in[0];
  const int*   mask  = (const int*)d_in[1];
  const float* wq_w  = (const float*)d_in[2];
  const float* wq_b  = (const float*)d_in[3];
  const float* wk_w  = (const float*)d_in[4];
  const float* wk_b  = (const float*)d_in[5];
  const float* wv_w  = (const float*)d_in[6];
  const float* wv_b  = (const float*)d_in[7];
  const float* wo_w  = (const float*)d_in[8];
  const float* wo_b  = (const float*)d_in[9];
  const float* ff1_w = (const float*)d_in[10];
  const float* ff1_b = (const float*)d_in[11];
  const float* ff2_w = (const float*)d_in[12];
  const float* ff2_b = (const float*)d_in[13];
  const float* ln1_a = (const float*)d_in[14];
  const float* ln1_b = (const float*)d_in[15];
  const float* ln2_a = (const float*)d_in[16];
  const float* ln2_b = (const float*)d_in[17];
  (void)in_sizes; (void)n_in; (void)out_size; (void)ws_size;

  char* ws = (char*)d_ws;
  size_t off = 0;
  auto alloc = [&](size_t bytes) -> void* {
    void* p = ws + off;
    off += (bytes + 255) & ~(size_t)255;
    return p;
  };
  bf16*  h1  = (bf16*)alloc((size_t)M_ * D_ * 2);
  bf16*  h2  = (bf16*)alloc((size_t)M_ * D_ * 2);
  bf16*  Qm  = (bf16*)alloc((size_t)M_ * D_ * 2);
  bf16*  Km  = (bf16*)alloc((size_t)M_ * D_ * 2);
  bf16*  Vm  = (bf16*)alloc((size_t)M_ * D_ * 2);
  bf16*  Om  = (bf16*)alloc((size_t)M_ * D_ * 2);
  float* x1  = (float*)alloc((size_t)M_ * D_ * 4);
  bf16*  a1  = (bf16*)alloc((size_t)M_ * FF_ * 2);
  bf16*  wqb = (bf16*)alloc((size_t)D_ * D_ * 2);
  bf16*  wkb = (bf16*)alloc((size_t)D_ * D_ * 2);
  bf16*  wvb = (bf16*)alloc((size_t)D_ * D_ * 2);
  bf16*  wob = (bf16*)alloc((size_t)D_ * D_ * 2);
  bf16*  f1b = (bf16*)alloc((size_t)FF_ * D_ * 2);
  bf16*  f2b = (bf16*)alloc((size_t)D_ * FF_ * 2);

  // weight down-conversion (recomputed every call; trivially bandwidth-bound)
  cvt_kernel<<<2048, 256, 0, stream>>>(wq_w,  wqb, D_ * D_);
  cvt_kernel<<<2048, 256, 0, stream>>>(wk_w,  wkb, D_ * D_);
  cvt_kernel<<<2048, 256, 0, stream>>>(wv_w,  wvb, D_ * D_);
  cvt_kernel<<<2048, 256, 0, stream>>>(wo_w,  wob, D_ * D_);
  cvt_kernel<<<4096, 256, 0, stream>>>(ff1_w, f1b, FF_ * D_);
  cvt_kernel<<<4096, 256, 0, stream>>>(ff2_w, f2b, D_ * FF_);

  // h1 = LN1(x)
  ln_kernel<<<M_, 256, 0, stream>>>(x, ln1_a, ln1_b, h1);

  const dim3 gD(D_ / 128, M_ / 128);   // 128x128 block tiles
  // Q/K/V projections (bf16 out)
  gemm_bf16_kernel<false, false, true><<<gD, 256, 0, stream>>>(h1, wqb, wq_b, nullptr, nullptr, Qm, M_, D_, D_);
  gemm_bf16_kernel<false, false, true><<<gD, 256, 0, stream>>>(h1, wkb, wk_b, nullptr, nullptr, Km, M_, D_, D_);
  gemm_bf16_kernel<false, false, true><<<gD, 256, 0, stream>>>(h1, wvb, wv_b, nullptr, nullptr, Vm, M_, D_, D_);

  // attention
  attn_kernel<<<dim3(B_ * H_, S_ / 128), 256, 0, stream>>>(Qm, Km, Vm, mask, Om);

  // x1 = x + O @ wo^T + b  (f32 out)
  gemm_bf16_kernel<false, true, false><<<gD, 256, 0, stream>>>(Om, wob, wo_b, x, x1, nullptr, M_, D_, D_);

  // h2 = LN2(x1)
  ln_kernel<<<M_, 256, 0, stream>>>(x1, ln2_a, ln2_b, h2);

  // a1 = relu(h2 @ ff1^T + b)  (bf16 out)
  const dim3 gF(FF_ / 128, M_ / 128);
  gemm_bf16_kernel<true, false, true><<<gF, 256, 0, stream>>>(h2, f1b, ff1_b, nullptr, nullptr, a1, M_, FF_, D_);

  // out = x1 + a1 @ ff2^T + b  (f32 out -> d_out)
  gemm_bf16_kernel<false, true, false><<<gD, 256, 0, stream>>>(a1, f2b, ff2_b, x1, (float*)d_out, nullptr, M_, D_, FF_);
}
